// Model_39676907883404
// MI455X (gfx1250) — matrix operand order repaired and checked by reading, not compiled, not run on hardware
//
#include <hip/hip_runtime.h>
#include <math.h>

typedef __bf16 bf16;
typedef __attribute__((ext_vector_type(16))) __bf16 v16bf;
typedef __attribute__((ext_vector_type(8)))  __bf16 v8bf;
typedef __attribute__((ext_vector_type(4)))  __bf16 v4bf;
typedef __attribute__((ext_vector_type(8)))  float  v8f;
typedef __attribute__((ext_vector_type(4)))  unsigned int v4u;
typedef __attribute__((ext_vector_type(8)))  int v8i;
typedef __attribute__((ext_vector_type(4)))  int v4i;

#ifndef __has_builtin
#define __has_builtin(x) 0
#endif

#if __has_builtin(__builtin_amdgcn_tensor_load_to_lds) && __has_builtin(__builtin_amdgcn_s_wait_tensorcnt)
#define USE_TDM 1
#else
#define USE_TDM 0
#endif

#if __has_include(<hip/amd_detail/amd_gfx1250_TDM.h>)
#define TDM_6ARG 1
#else
#define TDM_6ARG 0
#endif

union Frag { v16bf v; v8bf h[2]; };

#define BM 256
#define BN 128
#define BK 32
#define LDST 40   // LDS row stride in bf16 elems: 32 data + 8 pad (64B + 16B, TDM pad-compatible)

#if USE_TDM
// Issue a TDM 2D tile load: `rows` rows of BK bf16 elems, global row stride
// k_elems, into LDS at lds_addr with 16B padding every 64B (matches LDST=40).
__device__ __forceinline__ void tdm_load_2d(unsigned lds_addr, const bf16* gptr,
                                            int k_elems, int rows)
{
  unsigned long long ga = (unsigned long long)(const void*)gptr;
  v4u g0;
  g0[0] = 1u;                                               // count=1 (valid), user mode
  g0[1] = lds_addr;                                         // lds_addr [63:32]
  g0[2] = (unsigned)ga;                                     // global_addr[31:0]
  g0[3] = ((unsigned)(ga >> 32) & 0x01FFFFFFu) | 0x80000000u; // addr[56:32] | type=2
  const unsigned td0 = (unsigned)k_elems;                   // tensor_dim0 (in-bounds anyway)
  const unsigned td1 = 0x00100000u;                         // tensor_dim1: large
  v8i g1;
  g1[0] = (int)((1u << 16)      // data_size = 2 bytes
              | (1u << 20)      // pad_enable
              | (3u << 22)      // pad_interval: 16 DWORDs (64B)
              | (3u << 25));    // pad_amount: 4 DWORDs (16B)
  g1[1] = (int)((td0 & 0xFFFFu) << 16);                     // tensor_dim0 lo16 at [63:48]
  g1[2] = (int)((td0 >> 16) | ((td1 & 0xFFFFu) << 16));     // td0 hi16 | td1 lo16
  g1[3] = (int)(((td1 >> 16) & 0xFFFFu) | ((unsigned)BK << 16)); // td1 hi16 | tile_dim0
  g1[4] = rows;                                             // tile_dim1 | tile_dim2=0
  g1[5] = k_elems;                                          // tensor_dim0_stride lo32
  g1[6] = 0;
  g1[7] = 0;
  v4i z4 = {0, 0, 0, 0};
#if TDM_6ARG
  v8i z8 = {0, 0, 0, 0, 0, 0, 0, 0};
  __builtin_amdgcn_tensor_load_to_lds(g0, g1, z4, z4, z8, 0);
#else
  __builtin_amdgcn_tensor_load_to_lds(g0, g1, z4, z4, 0);
#endif
}
#endif

// C[m][n] = sum_k A[m][k] * Bt[n][k]  (+ bias[n])
// A: [M][K] row-major bf16, Bt: [N][K] row-major bf16.
template<bool OUT_BF16, bool HAS_BIAS>
__global__ __launch_bounds__(256) void gemm_wmma(
    const bf16* __restrict__ Av, const bf16* __restrict__ Btv,
    const float* __restrict__ bias, void* __restrict__ Cv,
    int M, int N, int K,
    long long sA, long long sB, long long sC)
{
  __shared__ bf16 As[2][BM * LDST];
  __shared__ bf16 Bs[2][BN * LDST];

  const int bz = blockIdx.z;
  const int blockN0 = blockIdx.x * BN;
  const int blockM0 = blockIdx.y * BM;
  const bf16* Atile = Av + (size_t)bz * sA + (size_t)blockM0 * K;
  const bf16* Btile = Btv + (size_t)bz * sB + (size_t)blockN0 * K;

  const int t    = threadIdx.x;
  const int wid  = t >> 5;
  const int lane = t & 31;
  const int lm   = lane & 15;
  const int lh   = lane >> 4;
  const int wm0  = (wid & 3) * 64;   // 4 waves tile M (64 rows each)
  const int wn0  = (wid >> 2) * 64;  // 2 waves tile N (64 cols each)

  v8f acc[4][4];
  for (int i = 0; i < 4; ++i)
    for (int j = 0; j < 4; ++j)
      for (int e = 0; e < 8; ++e) acc[i][j][e] = 0.0f;

  const int nkt = K / BK;

#if USE_TDM
  const unsigned ldsA0 = (unsigned)(unsigned long long)(const void*)&As[0][0];
  const unsigned ldsA1 = (unsigned)(unsigned long long)(const void*)&As[1][0];
  const unsigned ldsB0 = (unsigned)(unsigned long long)(const void*)&Bs[0][0];
  const unsigned ldsB1 = (unsigned)(unsigned long long)(const void*)&Bs[1][0];
  if (wid == 0) {
    tdm_load_2d(ldsA0, Atile, K, BM);
    tdm_load_2d(ldsB0, Btile, K, BN);
  }
#endif

  for (int kt = 0; kt < nkt; ++kt) {
    const int k0 = kt * BK;
#if USE_TDM
    const int cur = kt & 1;
    __syncthreads();   // all waves done reading buffer (cur^1) from prev iter
    if (wid == 0) {
      if (kt + 1 < nkt) {
        tdm_load_2d(cur ? ldsA0 : ldsA1, Atile + (size_t)(k0 + BK), K, BM);
        tdm_load_2d(cur ? ldsB0 : ldsB1, Btile + (size_t)(k0 + BK), K, BN);
        __builtin_amdgcn_s_wait_tensorcnt(2);  // current buffer's 2 DMAs done
      } else {
        __builtin_amdgcn_s_wait_tensorcnt(0);
      }
    }
    __syncthreads();   // current buffer visible to all waves
    const bf16* Asc = As[cur];
    const bf16* Bsc = Bs[cur];
#else
    // VGPR staging fallback
    {
      const v8bf* s = (const v8bf*)(Atile + (size_t)t * K + k0);
      v8bf* d = (v8bf*)&As[0][t * LDST];
      d[0] = s[0]; d[1] = s[1]; d[2] = s[2]; d[3] = s[3];
      const int brow = t >> 1, bkc = (t & 1) * 16;
      const v8bf* sb = (const v8bf*)(Btile + (size_t)brow * K + k0 + bkc);
      *(v8bf*)&Bs[0][brow * LDST + bkc]     = sb[0];
      *(v8bf*)&Bs[0][brow * LDST + bkc + 8] = sb[1];
    }
    __syncthreads();
    const bf16* Asc = As[0];
    const bf16* Bsc = Bs[0];
#endif

    Frag a[4], bb[4];
    const int ka0 = lh ? 8 : 0;   // A: lanes 0-15 K{0-7,16-23}; lanes 16-31 K{8-15,24-31}
    for (int tm = 0; tm < 4; ++tm) {
      const bf16* p = &Asc[(wm0 + tm * 16 + lm) * LDST];
      a[tm].h[0] = *(const v8bf*)(p + ka0);
      a[tm].h[1] = *(const v8bf*)(p + ka0 + 16);
    }
    const int kb0 = lh ? 8 : 0;   // B: lanes 0-15 K{0-7,16-23}; lanes 16-31 K{8-15,24-31}
    for (int tn = 0; tn < 4; ++tn) {
      const bf16* p = &Bsc[(wn0 + tn * 16 + lm) * LDST];
      bb[tn].h[0] = *(const v8bf*)(p + kb0);
      bb[tn].h[1] = *(const v8bf*)(p + kb0 + 16);
    }

    for (int tm = 0; tm < 4; ++tm)
      for (int tn = 0; tn < 4; ++tn)
        acc[tm][tn] = __builtin_amdgcn_wmma_f32_16x16x32_bf16(
            false, a[tm].v, false, bb[tn].v, (short)0, acc[tm][tn], false, false);
#if !USE_TDM
    __syncthreads();
#endif
  }

  // epilogue: C layout VGPR r -> M = r + 8*lh, N = lm
  for (int tm = 0; tm < 4; ++tm) {
    for (int tn = 0; tn < 4; ++tn) {
      const int n = blockN0 + wn0 + tn * 16 + lm;
      const float bvv = HAS_BIAS ? bias[n] : 0.0f;
      for (int r = 0; r < 8; ++r) {
        const int m = blockM0 + wm0 + tm * 16 + lh * 8 + r;
        const float val = acc[tm][tn][r] + bvv;
        if (OUT_BF16) {
          bf16* C = (bf16*)Cv + (size_t)bz * sC;
          C[(size_t)m * N + n] = (bf16)val;
        } else {
          float* C = (float*)Cv + (size_t)bz * sC;
          C[(size_t)m * N + n] = val;
        }
      }
    }
  }
}

// out[i] = (bf16) in[i], vectorized x4
__global__ __launch_bounds__(256) void convert_bf16_kernel(
    const float* __restrict__ in, bf16* __restrict__ out, int n4)
{
  int i = blockIdx.x * 256 + threadIdx.x;
  if (i >= n4) return;
  float4 f = ((const float4*)in)[i];
  v4bf o;
  o[0] = (bf16)f.x; o[1] = (bf16)f.y; o[2] = (bf16)f.z; o[3] = (bf16)f.w;
  *(v4bf*)(out + (size_t)i * 4) = o;
}

// Wt[n*K + k] = (bf16) W[k*N + n]
__global__ __launch_bounds__(256) void transpose_convert_kernel(
    const float* __restrict__ W, bf16* __restrict__ Wt, int K, int N)
{
  int tid = blockIdx.x * 256 + threadIdx.x;
  if (tid >= K * N) return;
  int k = tid / N, n = tid % N;
  Wt[(size_t)n * K + k] = (bf16)W[(size_t)k * N + n];
}

// softmax over feature dim (1024) per row; fp32 in -> bf16 out (row-major)
__global__ __launch_bounds__(256) void row_softmax_kernel(
    const float* __restrict__ in, bf16* __restrict__ out)
{
  const int D = 1024;
  __shared__ float red[256];
  const int t = threadIdx.x;
  const float* rp = in + (size_t)blockIdx.x * D;
  float4 x = ((const float4*)rp)[t];
  float lmv = fmaxf(fmaxf(x.x, x.y), fmaxf(x.z, x.w));
  red[t] = lmv; __syncthreads();
  for (int s = 128; s > 0; s >>= 1) { if (t < s) red[t] = fmaxf(red[t], red[t + s]); __syncthreads(); }
  float mx = red[0]; __syncthreads();
  float e0 = __expf(x.x - mx), e1 = __expf(x.y - mx), e2 = __expf(x.z - mx), e3 = __expf(x.w - mx);
  red[t] = e0 + e1 + e2 + e3; __syncthreads();
  for (int s = 128; s > 0; s >>= 1) { if (t < s) red[t] += red[t + s]; __syncthreads(); }
  float inv = 1.0f / red[0];
  bf16* op = out + (size_t)blockIdx.x * D + (size_t)t * 4;
  op[0] = (bf16)(e0 * inv); op[1] = (bf16)(e1 * inv);
  op[2] = (bf16)(e2 * inv); op[3] = (bf16)(e3 * inv);
}

// softmax over sequence dim L per (batch, feature); writes TRANSPOSED bf16 [B][D][L]
__global__ __launch_bounds__(256) void col_softmax_t_kernel(
    const float* __restrict__ in, bf16* __restrict__ outT, int L, int D)
{
  const int b = blockIdx.y;
  const int d = blockIdx.x * 256 + threadIdx.x;
  const float* base = in + (size_t)b * L * D + d;
  float mx = -3.4e38f;
  for (int l = 0; l < L; ++l) mx = fmaxf(mx, base[(size_t)l * D]);
  float s = 0.0f;
  for (int l = 0; l < L; ++l) s += __expf(base[(size_t)l * D] - mx);
  const float inv = 1.0f / s;
  bf16* o = outT + ((size_t)b * D + d) * L;
  for (int l = 0; l < L; ++l) o[l] = (bf16)(__expf(base[(size_t)l * D] - mx) * inv);
}

extern "C" void kernel_launch(void* const* d_in, const int* in_sizes, int n_in,
                              void* d_out, int out_size, void* d_ws, size_t ws_size,
                              hipStream_t stream)
{
  (void)in_sizes; (void)n_in; (void)out_size; (void)ws_size;
  const int B = 4, L = 4096, D = 1024;

  const float* q  = (const float*)d_in[0];
  const float* k  = (const float*)d_in[1];
  const float* v  = (const float*)d_in[2];
  const float* Wq = (const float*)d_in[3];
  const float* bq = (const float*)d_in[4];
  const float* Wk = (const float*)d_in[5];
  const float* bk = (const float*)d_in[6];
  const float* Wv = (const float*)d_in[7];
  const float* bv = (const float*)d_in[8];
  const float* Wo = (const float*)d_in[9];
  const float* bo = (const float*)d_in[10];
  float* out = (float*)d_out;

  char* ws = (char*)d_ws;
  size_t off = 0;
  float* tmp = (float*)(ws + off); off += (size_t)B * L * D * 4;  // fp32 pre-softmax
  bf16* x16  = (bf16*)(ws + off);  off += (size_t)B * L * D * 2;  // bf16 input (reused q/k/v)
  bf16* qp   = (bf16*)(ws + off);  off += (size_t)B * L * D * 2;  // [B*L][D]
  bf16* kpT  = (bf16*)(ws + off);  off += (size_t)B * L * D * 2;  // [B][D][L]
  bf16* vpT  = (bf16*)(ws + off);  off += (size_t)B * L * D * 2;  // [B][D][L]
  bf16* Mt   = (bf16*)(ws + off);  off += (size_t)B * D * D * 2;  // [B][d2][d1] = M^T
  bf16* ctx  = (bf16*)(ws + off);  off += (size_t)B * L * D * 2;  // [B*L][D]
  bf16* WqT  = (bf16*)(ws + off);  off += (size_t)D * D * 2;
  bf16* WkT  = (bf16*)(ws + off);  off += (size_t)D * D * 2;
  bf16* WvT  = (bf16*)(ws + off);  off += (size_t)D * D * 2;
  bf16* WoT  = (bf16*)(ws + off);  off += (size_t)D * D * 2;

  dim3 blk(256);
  const int nw = D * D;
  transpose_convert_kernel<<<(nw + 255) / 256, blk, 0, stream>>>(Wq, WqT, D, D);
  transpose_convert_kernel<<<(nw + 255) / 256, blk, 0, stream>>>(Wk, WkT, D, D);
  transpose_convert_kernel<<<(nw + 255) / 256, blk, 0, stream>>>(Wv, WvT, D, D);
  transpose_convert_kernel<<<(nw + 255) / 256, blk, 0, stream>>>(Wo, WoT, D, D);

  const int n4 = B * L * D / 4;
  dim3 gProj(D / BN, (B * L) / BM, 1);

  // qp = softmax_feat(q@Wq + bq)
  convert_bf16_kernel<<<(n4 + 255) / 256, blk, 0, stream>>>(q, x16, n4);
  gemm_wmma<false, true><<<gProj, blk, 0, stream>>>(x16, WqT, bq, tmp, B * L, D, D, 0, 0, 0);
  row_softmax_kernel<<<B * L, blk, 0, stream>>>(tmp, qp);

  // kpT = softmax_seq(k@Wk + bk)^T
  convert_bf16_kernel<<<(n4 + 255) / 256, blk, 0, stream>>>(k, x16, n4);
  gemm_wmma<false, true><<<gProj, blk, 0, stream>>>(x16, WkT, bk, tmp, B * L, D, D, 0, 0, 0);
  col_softmax_t_kernel<<<dim3(D / 256, B), blk, 0, stream>>>(tmp, kpT, L, D);

  // vpT = softmax_seq(v@Wv + bv)^T
  convert_bf16_kernel<<<(n4 + 255) / 256, blk, 0, stream>>>(v, x16, n4);
  gemm_wmma<false, true><<<gProj, blk, 0, stream>>>(x16, WvT, bv, tmp, B * L, D, D, 0, 0, 0);
  col_softmax_t_kernel<<<dim3(D / 256, B), blk, 0, stream>>>(tmp, vpT, L, D);

  // Mt[b][d2][d1] = sum_l vp[l][d2]*kp[l][d1]   (A=vpT, Bt=kpT, K=L)
  gemm_wmma<true, false><<<dim3(D / BN, D / BM, B), blk, 0, stream>>>(
      vpT, kpT, nullptr, Mt, D, D, L,
      (long long)D * L, (long long)D * L, (long long)D * D);

  // ctx[b] = qp[b] @ M[b]   (Bt = Mt = M^T in [N][K] layout)
  gemm_wmma<true, false><<<dim3(D / BN, L / BM, B), blk, 0, stream>>>(
      qp, Mt, nullptr, ctx, L, D, D,
      (long long)L * D, (long long)D * D, (long long)L * D);

  // out = ctx @ Wo + bo  (fp32 output)
  gemm_wmma<false, true><<<gProj, blk, 0, stream>>>(
      ctx, WoT, bo, out, B * L, D, D, 0, 0, 0);
}
